// DGN_MLPAgent_10849087390159
// MI455X (gfx1250) — matrix operand order repaired and checked by reading, not compiled, not run on hardware
//
#include <hip/hip_runtime.h>
#include <hip/hip_bf16.h>

// ---------------------------------------------------------------------------
// DGN forward, fused for MI455X (gfx1250, wave32, WMMA bf16).
// B=16384 graphs of N=32 agents; H=64; DIN=275; A=14.
// One workgroup (256 thr = 8 wave32) per graph; all intermediates in LDS.
// Weights pre-packed into WMMA B-fragment order (one 32 B load per lane per
// fragment); LDS strides padded for 64-bank conflict avoidance.
// ---------------------------------------------------------------------------

#define BATCH 16384
#define N_AG  32
#define DIN_  275
#define H_DIM 64
#define A_DIM 14
#define K_ENC 288            // DIN padded up to multiple of 32 (zero-filled)

// padded LDS row strides (elements); all keep 16 B row alignment and give
// mod-64-dword bank spread across the 16 rows a fragment load touches
#define XS 296               // xb stride  (148 dwords)
#define HS 72                // hb/qs/ks   (36 dwords)
#define VS 40                // vT         (20 dwords)
#define AS 40                // att        (20 dwords)

typedef __attribute__((ext_vector_type(16))) __bf16 v16bf;
typedef __attribute__((ext_vector_type(8)))  __bf16 v8bf;
typedef __attribute__((ext_vector_type(8)))  float  v8f;

// ws (bf16) layout: weights as packed 512-elem fragment blocks per
// (K-chunk, N-tile): block[lane*16 + e] = W[kc + (lane>>4)*16 + e][n0 + (lane&15)]
#define WS_WENC 0                              // [9 kc][4 nt] blocks
#define WS_PROJ (K_ENC * H_DIM)                // 6 x [2 kc][4 nt] blocks
#define WS_WQN  (WS_PROJ + 6 * H_DIM * H_DIM)  // [2 kc][1 nt] blocks (N pad 16)
#define WS_ELEMS (WS_WQN + H_DIM * 16)         // 44032 bf16 = 88 KB

__device__ __forceinline__ v8f wmma_bf16(v16bf a, v16bf b, v8f c) {
  return __builtin_amdgcn_wmma_f32_16x16x32_bf16(
      /*neg_a=*/false, a, /*neg_b=*/false, b,
      /*c_mod=*/(short)0, c, /*reuse_a=*/false, /*reuse_b=*/false);
}

// A fragment (16-bit A 16x32): per lane one contiguous 16xbf16 run:
//   K = kc + h*16 .. +15   (h = lane>>4)
__device__ __forceinline__ v16bf frag_a(const __bf16* src, int stride, int m0, int kc) {
  const int lane = threadIdx.x & 31;
  const int h    = lane >> 4;
  const int row  = m0 + (lane & 15);
  const __bf16* p = src + row * stride + kc + h * 16;
  v8bf lo = *(const v8bf*)p;          // 16 B aligned
  v8bf hi = *(const v8bf*)(p + 8);    // 16 B aligned
  v16bf f;
#pragma unroll
  for (int i = 0; i < 8; ++i) { f[i] = lo[i]; f[8 + i] = hi[i]; }
  return f;
}

// B fragment where B = src^T, src row-major [N][stride]:
//   lane: n = n0 + (lane&15);  f[e] = src[n][kc + h*16 + e]  (two 16 B runs)
__device__ __forceinline__ v16bf frag_bt(const __bf16* src, int stride, int n0, int kc) {
  const int lane = threadIdx.x & 31;
  const int h    = lane >> 4;
  const int row  = n0 + (lane & 15);
  const __bf16* p = src + row * stride + kc + h * 16;
  v8bf lo = *(const v8bf*)p;
  v8bf hi = *(const v8bf*)(p + 8);
  v16bf f;
#pragma unroll
  for (int i = 0; i < 8; ++i) { f[i] = lo[i]; f[8 + i] = hi[i]; }
  return f;
}

// Pre-packed weight fragment: one contiguous 32 B load per lane (global/L2).
__device__ __forceinline__ v16bf frag_w(const __bf16* wbase, int ntiles, int kc, int n0) {
  const int lane = threadIdx.x & 31;
  const __bf16* p = wbase + ((((kc >> 5) * ntiles) + (n0 >> 4)) << 9) + lane * 16;
  return *(const v16bf*)p;
}

// ---------------------------------------------------------------------------
// Prologue: convert fp32 weights -> bf16 fragment-packed workspace.
// ---------------------------------------------------------------------------
__global__ __launch_bounds__(256) void cvt_weights_kernel(
    const float* __restrict__ W_enc, const float* __restrict__ Wv1,
    const float* __restrict__ Wk1,   const float* __restrict__ Wq1,
    const float* __restrict__ Wv2,   const float* __restrict__ Wk2,
    const float* __restrict__ Wq2,   const float* __restrict__ Wqn,
    __bf16* __restrict__ ws) {
  int i = blockIdx.x * blockDim.x + threadIdx.x;
  if (i >= WS_ELEMS) return;

  if (i < WS_PROJ) {                         // W_enc: [288][64], NT=4
    int blk = i >> 9, within = i & 511;
    int lane = within >> 4, e = within & 15;
    int kc = (blk >> 2) << 5;                // NT = 4
    int n  = ((blk & 3) << 4) | (lane & 15);
    int k  = kc + ((lane >> 4) << 4) + e;
    ws[i] = (k < DIN_) ? (__bf16)W_enc[k * H_DIM + n] : (__bf16)0.0f;
  } else if (i < WS_WQN) {                   // six [64][64], NT=4 each
    int j = i - WS_PROJ;
    int s = j >> 12, t = j & 4095;           // 4096 elems per matrix
    const float* w;
    switch (s) {
      case 0: w = Wv1; break; case 1: w = Wk1; break; case 2: w = Wq1; break;
      case 3: w = Wv2; break; case 4: w = Wk2; break; default: w = Wq2; break;
    }
    int blk = t >> 9, within = t & 511;
    int lane = within >> 4, e = within & 15;
    int kc = (blk >> 2) << 5;
    int n  = ((blk & 3) << 4) | (lane & 15);
    int k  = kc + ((lane >> 4) << 4) + e;
    ws[i] = (__bf16)w[k * H_DIM + n];
  } else {                                   // Wqn: [64][14->16], NT=1
    int t = i - WS_WQN;
    int blk = t >> 9, within = t & 511;
    int lane = within >> 4, e = within & 15;
    int kc = blk << 5;
    int n  = lane & 15;
    int k  = kc + ((lane >> 4) << 4) + e;
    ws[i] = (n < A_DIM) ? (__bf16)Wqn[k * A_DIM + n] : (__bf16)0.0f;
  }
}

// ---------------------------------------------------------------------------
// Fused DGN forward: encoder -> attn1 -> attn2 -> head, one block per graph.
// ---------------------------------------------------------------------------
__global__ __launch_bounds__(256) void dgn_fused_kernel(
    const float* __restrict__ X,    const float* __restrict__ adj,
    const float* __restrict__ b_enc,
    const float* __restrict__ bv1, const float* __restrict__ bk1, const float* __restrict__ bq1,
    const float* __restrict__ bv2, const float* __restrict__ bk2, const float* __restrict__ bq2,
    const float* __restrict__ bqn,
    const __bf16* __restrict__ ws, float* __restrict__ out) {
  __shared__ __align__(32) __bf16 xb[N_AG * XS];      // encoder input (bf16)
  __shared__ __align__(32) __bf16 hb[N_AG * HS];      // layer input / output
  __shared__ __align__(32) __bf16 qs[N_AG * HS];      // q (row-major)
  __shared__ __align__(32) __bf16 ks[N_AG * HS];      // k (row-major, as B^T)
  __shared__ __align__(32) __bf16 vT[H_DIM * VS];     // v transposed [h][agent]
  __shared__ __align__(32) float  sc[N_AG * 33];      // scores fp32 (padded)
  __shared__ __align__(32) __bf16 att[N_AG * AS];     // softmax probs

  const int b    = blockIdx.x;
  const int tid  = threadIdx.x;
  const int w    = tid >> 5;                // wave id 0..7
  const int lane = tid & 31;
  const int half = lane >> 4;
  const int ln   = lane & 15;
  const int mt   = (w >> 2) * 16;           // row-tile of the 32x64 outputs
  const int nt   = (w & 3) * 16;            // col-tile

  // ---- stage 0: load X[b] -> LDS as bf16, zero-pad K to 288 -------------
  const float* Xb = X + (size_t)b * N_AG * DIN_;
  for (int i = tid; i < N_AG * K_ENC; i += 256) {
    int r = i / K_ENC, c = i - r * K_ENC;
    xb[r * XS + c] = (c < DIN_) ? (__bf16)Xb[r * DIN_ + c] : (__bf16)0.0f;
  }
  __syncthreads();

  // ---- stage 1: h1 = relu(X @ W_enc + b_enc), one 16x16 tile per wave ----
  {
    v8f acc = {0.f, 0.f, 0.f, 0.f, 0.f, 0.f, 0.f, 0.f};
    for (int kc = 0; kc < K_ENC; kc += 32) {
      v16bf a  = frag_a(xb, XS, mt, kc);
      v16bf bm = frag_w(ws + WS_WENC, 4, kc, nt);
      acc = wmma_bf16(a, bm, acc);
    }
    float bias = b_enc[nt + ln];
#pragma unroll
    for (int r = 0; r < 8; ++r) {
      int m = mt + r + 8 * half;
      hb[m * HS + nt + ln] = (__bf16)fmaxf(acc[r] + bias, 0.0f);
    }
  }
  __syncthreads();

  // ---- stages 2..3: two attention layers --------------------------------
  const float* adjb = adj + (size_t)b * N_AG * N_AG;
  for (int layer = 0; layer < 2; ++layer) {
    const __bf16* Wv = ws + WS_PROJ + layer * 3 * H_DIM * H_DIM;
    const __bf16* Wk = Wv + H_DIM * H_DIM;
    const __bf16* Wq = Wk + H_DIM * H_DIM;
    const float* bv = layer ? bv2 : bv1;
    const float* bk = layer ? bk2 : bk1;
    const float* bq = layer ? bq2 : bq1;

    // projections: q,k,v = relu(h @ W + b); each wave owns tile (mt,nt) x3
    {
      v8f cq = {0.f,0.f,0.f,0.f,0.f,0.f,0.f,0.f};
      v8f ck = cq, cv = cq;
      for (int kc = 0; kc < H_DIM; kc += 32) {
        v16bf a = frag_a(hb, HS, mt, kc);
        cq = wmma_bf16(a, frag_w(Wq, 4, kc, nt), cq);
        ck = wmma_bf16(a, frag_w(Wk, 4, kc, nt), ck);
        cv = wmma_bf16(a, frag_w(Wv, 4, kc, nt), cv);
      }
      float biq = bq[nt + ln], bik = bk[nt + ln], biv = bv[nt + ln];
#pragma unroll
      for (int r = 0; r < 8; ++r) {
        int m = mt + r + 8 * half;
        qs[m * HS + nt + ln] = (__bf16)fmaxf(cq[r] + biq, 0.0f);
        ks[m * HS + nt + ln] = (__bf16)fmaxf(ck[r] + bik, 0.0f);
        vT[(nt + ln) * VS + m] = (__bf16)fmaxf(cv[r] + biv, 0.0f);
      }
    }
    __syncthreads();

    // scores = q @ k^T  (32x32 = 4 tiles, waves 0..3; K = 64)
    if (w < 4) {
      int mt2 = (w >> 1) * 16, nt2 = (w & 1) * 16;
      v8f acc = {0.f, 0.f, 0.f, 0.f, 0.f, 0.f, 0.f, 0.f};
      for (int kc = 0; kc < H_DIM; kc += 32) {
        v16bf a  = frag_a(qs, HS, mt2, kc);
        v16bf bm = frag_bt(ks, HS, nt2, kc);      // B = k^T
        acc = wmma_bf16(a, bm, acc);
      }
#pragma unroll
      for (int r = 0; r < 8; ++r)
        sc[(mt2 + r + 8 * half) * 33 + nt2 + ln] = acc[r];
    }
    __syncthreads();

    // masked softmax over axis=2 (row length 32 == one wave32)
#pragma unroll
    for (int rr = 0; rr < 4; ++rr) {
      int row = w + 8 * rr;
      float msk = adjb[row * N_AG + lane];
      float s = sc[row * 33 + lane] * msk - 9.0e15f * (1.0f - msk);
      float mx = s;
      for (int o = 16; o > 0; o >>= 1) mx = fmaxf(mx, __shfl_xor(mx, o, 32));
      float e = __expf(s - mx);
      float sum = e;
      for (int o = 16; o > 0; o >>= 1) sum += __shfl_xor(sum, o, 32);
      att[row * AS + lane] = (__bf16)(e / sum);
    }
    __syncthreads();

    // h = att @ v  (K = 32, single chunk; one tile per wave; B = (vT)^T = v)
    {
      v16bf a  = frag_a(att, AS, mt, 0);
      v16bf bm = frag_bt(vT, VS, nt, 0);
      v8f acc  = {0.f, 0.f, 0.f, 0.f, 0.f, 0.f, 0.f, 0.f};
      acc = wmma_bf16(a, bm, acc);
      __syncthreads();                       // all reads of hb/att/vT done
#pragma unroll
      for (int r = 0; r < 8; ++r) {
        int m = mt + r + 8 * half;
        hb[m * HS + nt + ln] = (__bf16)acc[r];
      }
    }
    __syncthreads();
  }

  // ---- stage 4: q_out = h3 @ Wqn + bqn  (N padded to 16; waves 0,1) ------
  if (w < 2) {
    int mth = w * 16;
    v8f acc = {0.f, 0.f, 0.f, 0.f, 0.f, 0.f, 0.f, 0.f};
    for (int kc = 0; kc < H_DIM; kc += 32) {
      v16bf a  = frag_a(hb, HS, mth, kc);
      v16bf bm = frag_w(ws + WS_WQN, 1, kc, 0);
      acc = wmma_bf16(a, bm, acc);
    }
    if (ln < A_DIM) {
      float bias = bqn[ln];
#pragma unroll
      for (int r = 0; r < 8; ++r) {
        int m = mth + r + 8 * half;
        out[((size_t)b * N_AG + m) * A_DIM + ln] = acc[r] + bias;
      }
    }
  }
}

// ---------------------------------------------------------------------------
extern "C" void kernel_launch(void* const* d_in, const int* in_sizes, int n_in,
                              void* d_out, int out_size, void* d_ws, size_t ws_size,
                              hipStream_t stream) {
  const float* X     = (const float*)d_in[0];
  const float* hs    = (const float*)d_in[1];   // hidden_state (passthrough)
  const float* adj   = (const float*)d_in[2];
  const float* W_enc = (const float*)d_in[3];
  const float* b_enc = (const float*)d_in[4];
  const float* Wv1   = (const float*)d_in[5];
  const float* bv1   = (const float*)d_in[6];
  const float* Wk1   = (const float*)d_in[7];
  const float* bk1   = (const float*)d_in[8];
  const float* Wq1   = (const float*)d_in[9];
  const float* bq1   = (const float*)d_in[10];
  const float* Wv2   = (const float*)d_in[11];
  const float* bv2   = (const float*)d_in[12];
  const float* Wk2   = (const float*)d_in[13];
  const float* bk2   = (const float*)d_in[14];
  const float* Wq2   = (const float*)d_in[15];
  const float* bq2   = (const float*)d_in[16];
  const float* Wqn   = (const float*)d_in[17];
  const float* bqn   = (const float*)d_in[18];
  float*  out = (float*)d_out;
  __bf16* ws  = (__bf16*)d_ws;

  cvt_weights_kernel<<<(WS_ELEMS + 255) / 256, 256, 0, stream>>>(
      W_enc, Wv1, Wk1, Wq1, Wv2, Wk2, Wq2, Wqn, ws);

  dgn_fused_kernel<<<BATCH, 256, 0, stream>>>(
      X, adj, b_enc, bv1, bk1, bq1, bv2, bk2, bq2, bqn, ws, out);

  // outputs are (q, hidden_state) concatenated; hidden_state is passthrough
  hipMemcpyAsync(out + (size_t)BATCH * N_AG * A_DIM, hs,
                 (size_t)BATCH * N_AG * H_DIM * sizeof(float),
                 hipMemcpyDeviceToDevice, stream);
}
